// MultiHeadAttention_77008763617462
// MI455X (gfx1250) — matrix operand order repaired and checked by reading, not compiled, not run on hardware
//
#include <hip/hip_runtime.h>
#include <hip/hip_bf16.h>
#include <stdint.h>

// ---------------------------------------------------------------------------
// MultiHeadAttention (RealFormer residual scores) for MI455X / gfx1250
//   B=8, S=1024, D=1024, H=16, HD=64
// fp32 -> bf16 WMMA (v_wmma_f32_16x16x32_bf16), fp32 accumulate,
// flash-attention online softmax fused with +mask +prev_attn_weights.
// Transposed B-operand tiles are built at staging time with pair-packed
// ds_store_b32 scatter; all fragment reads are contiguous ds_load_b128.
// K tiles staged via async global->LDS; prev-attn stream prefetched.
// ---------------------------------------------------------------------------

typedef __attribute__((ext_vector_type(16))) __bf16 v16bf;
typedef __attribute__((ext_vector_type(2)))  __bf16 bf16x2;
typedef __attribute__((ext_vector_type(8)))  float  v8f;

#define BB 8
#define SS 1024
#define DD 1024
#define HH 16
#define HDIM 64

// fp32 -> bf16 bits via hardware convert (RNE)
static __device__ __forceinline__ uint16_t f2bf_bits(float f) {
    return __builtin_bit_cast(uint16_t, (__bf16)f);
}
// pack two fp32 into a dword of two bf16 (lo, hi)
static __device__ __forceinline__ uint32_t pack2(float lo, float hi) {
    return (uint32_t)f2bf_bits(lo) | ((uint32_t)f2bf_bits(hi) << 16);
}
// pack two u16 into a dword
static __device__ __forceinline__ uint32_t pack2u(uint16_t lo, uint16_t hi) {
    return (uint32_t)lo | ((uint32_t)hi << 16);
}
// load two adjacent bf16 (one dword) from LDS/global
static __device__ __forceinline__ bf16x2 ld_pair(const uint16_t* p) {
    return __builtin_bit_cast(bf16x2, *(const uint32_t*)p);
}
// async copy 16B global -> LDS (per lane), tracked by ASYNCcnt
static __device__ __forceinline__ void async_ld_b128(uint16_t* lds_dst,
                                                     const uint16_t* gsrc) {
    const uint32_t loff = (uint32_t)(uintptr_t)lds_dst;  // low 32 bits = LDS offset
    asm volatile("global_load_async_to_lds_b128 %0, %1, off"
                 :: "v"(loff), "v"(gsrc) : "memory");
}

// ===========================================================================
// Kernel 1: Y = X @ W + bias  (X: 8192x1024 fp32, W: 1024x1024 fp32)
// Output bf16, head-major layout [B, H, S, HD].
// Block = 128 threads (4 waves). Tile = 64(M) x 64(N), K-step = 64.
// ===========================================================================
__global__ __launch_bounds__(128)
void qkv_proj_kernel(const float* __restrict__ X, const float* __restrict__ W,
                     const float* __restrict__ bias, uint16_t* __restrict__ dst)
{
    __shared__ uint16_t Xl[64][72];   // [m][k]  natural
    __shared__ uint16_t Wt[64][72];   // [n][k]  transposed W tile

    const int tid  = threadIdx.x;
    const int wave = tid >> 5;
    const int lane = tid & 31;
    const int half = lane >> 4;
    const int l16  = lane & 15;
    const int m0   = blockIdx.x * 64;
    const int n0   = blockIdx.y * 64;

    v8f acc[4] = {};

    for (int k0 = 0; k0 < DD; k0 += 64) {
        // ---- stage X tile 64x64 fp32->bf16, packed b128 stores ----
        {
            const int row = tid >> 1;
            const int ch  = (tid & 1) * 32;
            const float* src = X + (size_t)(m0 + row) * DD + k0 + ch;
            uint32_t pk[16];
#pragma unroll
            for (int i = 0; i < 16; ++i) pk[i] = pack2(src[2 * i], src[2 * i + 1]);
            uint32_t* q = (uint32_t*)&Xl[row][ch];
#pragma unroll
            for (int i = 0; i < 4; ++i)
                ((uint4*)q)[i] = make_uint4(pk[4 * i], pk[4 * i + 1],
                                            pk[4 * i + 2], pk[4 * i + 3]);
        }
        // ---- stage W transposed: two k-rows pair-packed into b32 ----
        {
            const int kp   = tid & 31;            // k-row pair: rows 2kp, 2kp+1
            const int nseg = (tid >> 5) * 16;     // 16 n-columns
            const float* r0 = W + (size_t)(k0 + 2 * kp)     * DD + n0 + nseg;
            const float* r1 = W + (size_t)(k0 + 2 * kp + 1) * DD + n0 + nseg;
#pragma unroll
            for (int i = 0; i < 16; ++i)
                *(uint32_t*)&Wt[nseg + i][2 * kp] = pack2(r0[i], r1[i]);
        }
        __syncthreads();

#pragma unroll
        for (int c = 0; c < 2; ++c) {
            // A fragment: rows 16*wave..+15, K chunk 32c..32c+31 (contiguous)
            v16bf a;
            {
                const int m  = 16 * wave + l16;
                const int kb = 16 * half;
#pragma unroll
                for (int v = 0; v < 8; ++v) {
                    const int k = 32 * c + kb + 2 * v;
                    bf16x2 p = ld_pair(&Xl[m][k]);
                    a[2 * v] = p[0]; a[2 * v + 1] = p[1];
                }
            }
#pragma unroll
            for (int g = 0; g < 4; ++g) {
                // B fragment: contiguous b128 reads from transposed Wt[n][k]
                v16bf bfr;
                const int n  = 16 * g + l16;
                const int kb = 32 * c + 16 * half;
#pragma unroll
                for (int v = 0; v < 8; ++v) {
                    bf16x2 p = ld_pair(&Wt[n][kb + 2 * v]);
                    bfr[2 * v] = p[0]; bfr[2 * v + 1] = p[1];
                }
                acc[g] = __builtin_amdgcn_wmma_f32_16x16x32_bf16(
                    false, a, false, bfr, (short)0, acc[g], false, false);
            }
        }
        __syncthreads();
    }

    // ---- epilogue: +bias, fp32 -> bf16, store head-major [B,H,S,HD] ----
#pragma unroll
    for (int g = 0; g < 4; ++g) {
        const int n  = n0 + 16 * g + l16;
        const int h  = n >> 6;
        const int hd = n & 63;
        const float bv = bias[n];
#pragma unroll
        for (int r = 0; r < 8; ++r) {
            const int m = m0 + 16 * wave + r + 8 * half;
            const int b = m >> 10;
            const int s = m & 1023;
            dst[(((size_t)(b * HH + h) * SS + s)) * HDIM + hd] =
                f2bf_bits(acc[g][r] + bv);
        }
    }
}

// ===========================================================================
// Kernel 2: fused attention per (b, h, 64-row q-tile).
// Block = 128 threads (4 waves); wave w owns q rows [16w, 16w+16).
// Online softmax over 16 key tiles of 64.
// ===========================================================================
__global__ __launch_bounds__(128)
void attn_kernel(const uint16_t* __restrict__ Q, const uint16_t* __restrict__ K,
                 const uint16_t* __restrict__ V, const float* __restrict__ mask,
                 const float* __restrict__ prev, float* __restrict__ out)
{
    __shared__ uint16_t Kl[64][72];      // [key][hd]  async-staged natural
    __shared__ uint16_t Vt[64][72];      // [hd][key]  transposed V tile
    __shared__ uint16_t Pl[4][16][72];   // per-wave probs strip [m][key]

    const int tid  = threadIdx.x;
    const int wave = tid >> 5;
    const int lane = tid & 31;
    const int half = lane >> 4;
    const int l16  = lane & 15;

    const int qt = blockIdx.x;           // 0..15
    const int bh = blockIdx.y;           // 0..127
    const int b  = bh >> 4;
    const int h  = bh & 15;

    const size_t head_off = (size_t)bh * SS * HDIM;
    const int qs = qt * 64 + 16 * wave;  // wave strip base row

    // ---- Q fragments: 16x64 strip -> two A-frags (K chunks of 32) ----
    v16bf aQ[2];
    {
        const uint16_t* qptr = Q + head_off + (size_t)(qs + l16) * HDIM;
        const int kb = 16 * half;
#pragma unroll
        for (int c = 0; c < 2; ++c) {
#pragma unroll
            for (int v = 0; v < 8; ++v) {
                const int k = 32 * c + kb + 2 * v;
                bf16x2 p = ld_pair(&qptr[k]);
                aQ[c][2 * v] = p[0]; aQ[c][2 * v + 1] = p[1];
            }
        }
    }

    v8f o[4] = {};
    float mrow[8], lrow[8];
#pragma unroll
    for (int r = 0; r < 8; ++r) { mrow[r] = -1e30f; lrow[r] = 0.0f; }

    const float SCALE = 0.125f;          // HD^-0.5 = 1/8

    for (int kt = 0; kt < SS / 64; ++kt) {
        const int kb0 = kt * 64;
        // ---- stage K tile via async global->LDS (natural [key][hd]) ----
        {
            const uint16_t* kbase = K + head_off + (size_t)kb0 * HDIM;
#pragma unroll
            for (int i = 0; i < 4; ++i) {
                const int chunk = tid + 128 * i;        // 512 x 16B chunks
                const int row   = chunk >> 3;           // key
                const int col   = (chunk & 7) * 8;      // hd
                async_ld_b128(&Kl[row][col], kbase + chunk * 8);
            }
        }
        // ---- stage V transposed: two key-rows pair-packed into b32 ----
        {
            const int kp   = tid & 31;           // key pair: 2kp, 2kp+1
            const int hseg = (tid >> 5) * 16;    // 16 hd values
            const uint16_t* r0 = V + head_off + (size_t)(kb0 + 2 * kp)     * HDIM + hseg;
            const uint16_t* r1 = V + head_off + (size_t)(kb0 + 2 * kp + 1) * HDIM + hseg;
            uint16_t a0[16], a1[16];
#pragma unroll
            for (int i = 0; i < 8; ++i) {
                ((uint32_t*)a0)[i] = ((const uint32_t*)r0)[i];
                ((uint32_t*)a1)[i] = ((const uint32_t*)r1)[i];
            }
#pragma unroll
            for (int i = 0; i < 16; ++i)
                *(uint32_t*)&Vt[hseg + i][2 * kp] = pack2u(a0[i], a1[i]);
        }
        // ---- prefetch next key tile's prev-attn rows (HBM bottleneck) ----
        if (kt + 1 < SS / 64) {
            const int kcol = kb0 + 64 + l16 * 4;
#pragma unroll
            for (int r = 0; r < 8; ++r) {
                const int qrow = qs + r + 8 * half;
                __builtin_prefetch(&prev[((size_t)bh * SS + qrow) * SS + kcol], 0, 1);
            }
        }
        asm volatile("s_wait_asynccnt 0" ::: "memory");
        __syncthreads();

        // ---- S = Q K^T : 16 x 64, K-dim = HD = 64 ----
        v8f sacc[4] = {};
#pragma unroll
        for (int g = 0; g < 4; ++g) {
            const int n = 16 * g + l16;
#pragma unroll
            for (int c = 0; c < 2; ++c) {
                v16bf bk;                 // contiguous: transpose is inherent
                const int kb = 32 * c + 16 * half;
#pragma unroll
                for (int v = 0; v < 8; ++v) {
                    bf16x2 p = ld_pair(&Kl[n][kb + 2 * v]);
                    bk[2 * v] = p[0]; bk[2 * v + 1] = p[1];
                }
                sacc[g] = __builtin_amdgcn_wmma_f32_16x16x32_bf16(
                    false, aQ[c], false, bk, (short)0, sacc[g], false, false);
            }
        }

        // ---- scale + mask + prev_attn (the 512MB fp32 stream) ----
        float sv[4][8];
#pragma unroll
        for (int g = 0; g < 4; ++g) {
            const int kcol = kb0 + 16 * g + l16;
#pragma unroll
            for (int r = 0; r < 8; ++r) {
                const int qrow = qs + r + 8 * half;
                const float mv = mask[((size_t)b * SS + qrow) * SS + kcol];
                const float pv = prev[((size_t)bh * SS + qrow) * SS + kcol];
                sv[g][r] = sacc[g][r] * SCALE + mv + pv;
            }
        }

        // ---- online softmax (rows live across 16-lane groups) ----
#pragma unroll
        for (int r = 0; r < 8; ++r) {
            float mx = fmaxf(fmaxf(sv[0][r], sv[1][r]),
                             fmaxf(sv[2][r], sv[3][r]));
#pragma unroll
            for (int off = 8; off >= 1; off >>= 1)
                mx = fmaxf(mx, __shfl_xor(mx, off, 32));
            const float newm  = fmaxf(mrow[r], mx);
            const float alpha = __expf(mrow[r] - newm);
            float rs = 0.0f;
#pragma unroll
            for (int g = 0; g < 4; ++g) {
                const float p = __expf(sv[g][r] - newm);
                sv[g][r] = p;
                rs += p;
            }
#pragma unroll
            for (int off = 8; off >= 1; off >>= 1)
                rs += __shfl_xor(rs, off, 32);
            lrow[r] = lrow[r] * alpha + rs;
            mrow[r] = newm;
#pragma unroll
            for (int g = 0; g < 4; ++g) o[g][r] *= alpha;
        }

        // ---- C-layout -> A-layout via per-wave LDS region ----
#pragma unroll
        for (int g = 0; g < 4; ++g)
#pragma unroll
            for (int r = 0; r < 8; ++r)
                Pl[wave][r + 8 * half][16 * g + l16] = f2bf_bits(sv[g][r]);

        __builtin_amdgcn_wave_barrier();
        asm volatile("s_wait_dscnt 0" ::: "memory");

        v16bf aP[2];
        {
            const int kb = 16 * half;
#pragma unroll
            for (int c = 0; c < 2; ++c) {
#pragma unroll
                for (int v = 0; v < 8; ++v) {
                    const int k = 32 * c + kb + 2 * v;
                    bf16x2 p = ld_pair(&Pl[wave][l16][k]);
                    aP[c][2 * v] = p[0]; aP[c][2 * v + 1] = p[1];
                }
            }
        }

        // ---- O += P @ V : 16 x 64, K-dim = 64 keys ----
#pragma unroll
        for (int g = 0; g < 4; ++g) {
            const int n = 16 * g + l16;      // hd column
#pragma unroll
            for (int c = 0; c < 2; ++c) {
                v16bf bv;                    // contiguous from transposed Vt
                const int kb = 32 * c + 16 * half;
#pragma unroll
                for (int v = 0; v < 8; ++v) {
                    bf16x2 p = ld_pair(&Vt[n][kb + 2 * v]);
                    bv[2 * v] = p[0]; bv[2 * v + 1] = p[1];
                }
                o[g] = __builtin_amdgcn_wmma_f32_16x16x32_bf16(
                    false, aP[c], false, bv, (short)0, o[g], false, false);
            }
        }
        __syncthreads();
    }

    // ---- normalize by 1/l and store (B, S, D) fp32 ----
#pragma unroll
    for (int r = 0; r < 8; ++r) {
        const float inv = 1.0f / lrow[r];
        const int srow  = qs + r + 8 * half;
#pragma unroll
        for (int g = 0; g < 4; ++g) {
            const int d = h * HDIM + 16 * g + l16;
            out[((size_t)b * SS + srow) * DD + d] = o[g][r] * inv;
        }
    }
}

// ===========================================================================
extern "C" void kernel_launch(void* const* d_in, const int* in_sizes, int n_in,
                              void* d_out, int out_size, void* d_ws, size_t ws_size,
                              hipStream_t stream) {
    (void)in_sizes; (void)n_in; (void)out_size; (void)ws_size;
    const float* hidden = (const float*)d_in[0];
    const float* amask  = (const float*)d_in[1];
    const float* prev   = (const float*)d_in[2];
    const float* Wq     = (const float*)d_in[3];
    const float* bq     = (const float*)d_in[4];
    const float* Wk     = (const float*)d_in[5];
    const float* bk     = (const float*)d_in[6];
    const float* Wv     = (const float*)d_in[7];
    const float* bv     = (const float*)d_in[8];
    float* out = (float*)d_out;

    const size_t elems = (size_t)BB * SS * DD;       // 8M bf16 each
    uint16_t* q_ws = (uint16_t*)d_ws;
    uint16_t* k_ws = q_ws + elems;
    uint16_t* v_ws = k_ws + elems;

    dim3 gridP(BB * SS / 64, DD / 64);               // 128 x 16
    qkv_proj_kernel<<<gridP, 128, 0, stream>>>(hidden, Wq, bq, q_ws);
    qkv_proj_kernel<<<gridP, 128, 0, stream>>>(hidden, Wk, bk, k_ws);
    qkv_proj_kernel<<<gridP, 128, 0, stream>>>(hidden, Wv, bv, v_ws);

    dim3 gridA(SS / 64, BB * HH);                    // 16 x 128
    attn_kernel<<<gridA, 128, 0, stream>>>(q_ws, k_ws, v_ws, amask, prev, out);
}
